// NetGINE_v2_35459249995957
// MI455X (gfx1250) — matrix operand order repaired and checked by reading, not compiled, not run on hardware
//
#include <hip/hip_runtime.h>
#include <hip/hip_bf16.h>

// ---------------------------------------------------------------------------
// NetGINE (GINConv x6 + Set2Set + MLP head) for gfx1250 (MI455X).
// Edge/node MLP GEMMs use v_wmma_f32_16x16x32_f16 (f16 operands, f32 accum).
// Edge pipeline fully fused: bond-MLP -> gather x[src] -> relu -> atomic
// scatter-add, so only edge_attr streams from HBM (~205MB/layer).
// ---------------------------------------------------------------------------

typedef __attribute__((ext_vector_type(16))) _Float16 v16h;
typedef __attribute__((ext_vector_type(8)))  _Float16 v8h;
typedef __attribute__((ext_vector_type(8)))  float    v8f;
typedef __attribute__((ext_vector_type(4)))  float    v4f;

constexpr int NN  = 50000;   // nodes
constexpr int NE  = 800000;  // edges
constexpr int NG  = 256;     // graphs
constexpr int D   = 64;      // feature dim
constexpr int NL  = 6;       // GIN layers
constexpr int NC  = 12;      // classes
constexpr int S2S = 6;       // set2set steps
constexpr int NODE_TILES = NN / 16;  // 3125

#define WMMA_F16(a, b, c) \
  __builtin_amdgcn_wmma_f32_16x16x32_f16(false, (a), false, (b), (short)0, (c), false, false)

__device__ __forceinline__ void atomAdd(float* p, float v) {
  __hip_atomic_fetch_add(p, v, __ATOMIC_RELAXED, __HIP_MEMORY_SCOPE_AGENT);
}

// float -> order-preserving uint (for atomicMax-based segment max)
__device__ __forceinline__ unsigned fenc(float f) {
  unsigned u = __float_as_uint(f);
  return (u & 0x80000000u) ? ~u : (u | 0x80000000u);
}
__device__ __forceinline__ float fdec(unsigned u) {
  unsigned v = (u & 0x80000000u) ? (u & 0x7fffffffu) : ~u;
  return __uint_as_float(v);
}

// ---- WMMA fragment builders ------------------------------------------------
// 16-bit A 16x32 layout (ISA 7.12.2): lane L (row = L&15) holds two contiguous
// 8-element K chunks: [kb, kb+8) and [kb+16, kb+24), kb = kc*32 + (L>=16 ? 8:0).
__device__ __forceinline__ v16h a_frag_from_f32(const float* __restrict__ row, int kc, int hi) {
  int kb = kc * 32 + hi * 8;
  v4f x0 = *(const v4f*)(row + kb);
  v4f x1 = *(const v4f*)(row + kb + 4);
  v4f x2 = *(const v4f*)(row + kb + 16);
  v4f x3 = *(const v4f*)(row + kb + 20);
  v16h a;
#pragma unroll
  for (int i = 0; i < 4; ++i) {
    a[i]      = (_Float16)x0[i];
    a[4 + i]  = (_Float16)x1[i];
    a[8 + i]  = (_Float16)x2[i];
    a[12 + i] = (_Float16)x3[i];
  }
  return a;
}

// A fragment computed as (1+eps)*x + agg on the fly (node MLP input)
__device__ __forceinline__ v16h a_frag_from_h(const float* __restrict__ xr,
                                              const float* __restrict__ ar,
                                              float ep, int kc, int hi) {
  int kb = kc * 32 + hi * 8;
  v16h a;
#pragma unroll
  for (int i = 0; i < 8; ++i) {
    a[i]     = (_Float16)(ep * xr[kb + i]      + ar[kb + i]);
    a[8 + i] = (_Float16)(ep * xr[kb + 16 + i] + ar[kb + 16 + i]);
  }
  return a;
}

__device__ __forceinline__ v16h a_frag_from_lds(const _Float16* row, int kc, int hi) {
  int kb = kc * 32 + hi * 8;
  v8h lo = *(const v8h*)(row + kb);
  v8h hh = *(const v8h*)(row + kb + 16);
  return __builtin_shufflevector(lo, hh, 0, 1, 2, 3, 4, 5, 6, 7,
                                 8, 9, 10, 11, 12, 13, 14, 15);
}

// 16-bit B 32x16 layout: lane -> column N, same K chunks as A: [kb, kb+8) and [kb+16, kb+24).
// wt is stored column-major (wt[n*64 + k]) so this is two 16B loads.
__device__ __forceinline__ v16h b_frag(const _Float16* __restrict__ wt, int n, int kc, int hi) {
  const _Float16* p = wt + n * 64 + kc * 32 + hi * 8;
  v8h lo = *(const v8h*)(p);
  v8h hh = *(const v8h*)(p + 16);
  return __builtin_shufflevector(lo, hh, 0, 1, 2, 3, 4, 5, 6, 7,
                                 8, 9, 10, 11, 12, 13, 14, 15);
}

// ---- weight prep: f32 [l][k][n] -> f16 column-major [l][mat][n][k] ---------
__global__ void prep_weights(const float* __restrict__ bw1, const float* __restrict__ bw2,
                             const float* __restrict__ mw1, const float* __restrict__ mw2,
                             _Float16* __restrict__ wt) {
  int idx = blockIdx.x * 256 + threadIdx.x;  // NL*4*D*D = 98304
  if (idx >= NL * 4 * D * D) return;
  int k = idx & 63;
  int n = (idx >> 6) & 63;
  int mat = (idx >> 12) & 3;
  int l = idx >> 14;
  const float* srcs[4] = {bw1, bw2, mw1, mw2};
  float v = srcs[mat][(l * D + k) * D + n];
  wt[((l * 4 + mat) * D + n) * D + k] = (_Float16)v;
}

__global__ void zero_f4(float* __restrict__ p, int n4) {
  int i = blockIdx.x * 256 + threadIdx.x;
  if (i < n4) *(v4f*)(p + 4 * i) = (v4f){0.f, 0.f, 0.f, 0.f};
}

__global__ void zero_stats(unsigned* __restrict__ emax, float* __restrict__ asum,
                           float* __restrict__ rtmp) {
  int i = blockIdx.x * 256 + threadIdx.x;
  if (i < NG) { emax[i] = 0x007FFFFFu; /* == fenc(-inf) */ asum[i] = 0.f; }
  if (i < NG * D) rtmp[i] = 0.f;
}

// ---- fused edge pipeline ---------------------------------------------------
// per wave: 16 edges. ee = relu(ea@W1+b1)@W2+b2 ; msg = relu(x[src]+ee);
// atomic agg[dst] += msg.
__global__ void edge_kernel(const float* __restrict__ x, const float* __restrict__ ea,
                            const _Float16* __restrict__ wt1, const float* __restrict__ b1,
                            const _Float16* __restrict__ wt2, const float* __restrict__ b2,
                            const int* __restrict__ src, const int* __restrict__ dst,
                            float* __restrict__ agg) {
  __shared__ __align__(16) _Float16 hid[8][16 * D];
  int wave = threadIdx.x >> 5;
  int lane = threadIdx.x & 31;
  int tile = blockIdx.x * 8 + wave;
  int e0 = tile * 16;
  int row = lane & 15;
  int hi = (lane >> 4) & 1;
  int nl = lane & 15;

  const float* arow = ea + (size_t)(e0 + row) * D;
  // stream-ahead prefetch of edge_attr (speculative, OOB-safe)
  __builtin_prefetch(arow + (size_t)2048 * D, 0, 1);

  v16h a0 = a_frag_from_f32(arow, 0, hi);
  v16h a1 = a_frag_from_f32(arow, 1, hi);
  _Float16* hrow = &hid[wave][0];

#pragma unroll
  for (int nt = 0; nt < 4; ++nt) {
    int n = nt * 16 + nl;
    v16h bf0 = b_frag(wt1, n, 0, hi);
    v16h bf1 = b_frag(wt1, n, 1, hi);
    v8f acc = {};
    acc = WMMA_F16(a0, bf0, acc);
    acc = WMMA_F16(a1, bf1, acc);
    float bias = b1[n];
#pragma unroll
    for (int r = 0; r < 8; ++r) {
      float v = acc[r] + bias;
      v = v > 0.f ? v : 0.f;
      hrow[(r + hi * 8) * D + n] = (_Float16)v;  // D layout -> row-major LDS
    }
  }
  __syncthreads();

  v16h c0 = a_frag_from_lds(hrow + row * D, 0, hi);
  v16h c1 = a_frag_from_lds(hrow + row * D, 1, hi);

#pragma unroll
  for (int nt = 0; nt < 4; ++nt) {
    int n = nt * 16 + nl;
    v16h bf0 = b_frag(wt2, n, 0, hi);
    v16h bf1 = b_frag(wt2, n, 1, hi);
    v8f acc = {};
    acc = WMMA_F16(c0, bf0, acc);
    acc = WMMA_F16(c1, bf1, acc);
    float bias = b2[n];
#pragma unroll
    for (int r = 0; r < 8; ++r) {
      int e = e0 + r + hi * 8;
      int s = src[e];
      float msg = x[(size_t)s * D + n] + acc[r] + bias;
      msg = msg > 0.f ? msg : 0.f;
      atomAdd(&agg[(size_t)dst[e] * D + n], msg);
    }
  }
}

// ---- node MLP: x' = relu(relu(h@W1+b1)@W2+b2), h = (1+eps)x + agg ----------
__global__ void node_kernel(const float* __restrict__ xin, const float* __restrict__ agg,
                            const float* __restrict__ epsArr, int layer,
                            const _Float16* __restrict__ wt1, const float* __restrict__ b1,
                            const _Float16* __restrict__ wt2, const float* __restrict__ b2,
                            float* __restrict__ xout) {
  __shared__ __align__(16) _Float16 hid[8][16 * D];
  int wave = threadIdx.x >> 5;
  int lane = threadIdx.x & 31;
  int tile = blockIdx.x * 8 + wave;
  if (tile > NODE_TILES - 1) tile = NODE_TILES - 1;  // uniform clamp, keeps EXEC full
  int r0 = tile * 16;
  int row = lane & 15;
  int hi = (lane >> 4) & 1;
  int nl = lane & 15;

  float ep = 1.0f + epsArr[layer];
  const float* xr = xin + (size_t)(r0 + row) * D;
  const float* ar = agg + (size_t)(r0 + row) * D;
  v16h a0 = a_frag_from_h(xr, ar, ep, 0, hi);
  v16h a1 = a_frag_from_h(xr, ar, ep, 1, hi);
  _Float16* hrow = &hid[wave][0];

#pragma unroll
  for (int nt = 0; nt < 4; ++nt) {
    int n = nt * 16 + nl;
    v16h bf0 = b_frag(wt1, n, 0, hi);
    v16h bf1 = b_frag(wt1, n, 1, hi);
    v8f acc = {};
    acc = WMMA_F16(a0, bf0, acc);
    acc = WMMA_F16(a1, bf1, acc);
    float bias = b1[n];
#pragma unroll
    for (int r = 0; r < 8; ++r) {
      float v = acc[r] + bias;
      v = v > 0.f ? v : 0.f;
      hrow[(r + hi * 8) * D + n] = (_Float16)v;
    }
  }
  __syncthreads();

  v16h c0 = a_frag_from_lds(hrow + row * D, 0, hi);
  v16h c1 = a_frag_from_lds(hrow + row * D, 1, hi);

#pragma unroll
  for (int nt = 0; nt < 4; ++nt) {
    int n = nt * 16 + nl;
    v16h bf0 = b_frag(wt2, n, 0, hi);
    v16h bf1 = b_frag(wt2, n, 1, hi);
    v8f acc = {};
    acc = WMMA_F16(c0, bf0, acc);
    acc = WMMA_F16(c1, bf1, acc);
    float bias = b2[n];
#pragma unroll
    for (int r = 0; r < 8; ++r) {
      float v = acc[r] + bias;
      v = v > 0.f ? v : 0.f;
      xout[(size_t)(r0 + r + hi * 8) * D + n] = v;
    }
  }
}

// ---- Set2Set ---------------------------------------------------------------
__device__ __forceinline__ float sigmoidf(float x) { return 1.f / (1.f + expf(-x)); }

__global__ void gates_kernel(const float* __restrict__ qstar, const float* __restrict__ h,
                             const float* __restrict__ w_ih, const float* __restrict__ w_hh,
                             const float* __restrict__ b_ih, const float* __restrict__ b_hh,
                             float* __restrict__ gates) {
  __shared__ float qs[3 * D];  // [0,128): q_star row, [128,192): h row
  int b = blockIdx.x;
  int u = threadIdx.x;  // 256 = 4*D gate units
  if (u < 2 * D) qs[u] = qstar[b * 2 * D + u];
  else if (u < 3 * D) qs[u] = h[b * D + (u - 2 * D)];
  __syncthreads();
  float g = b_ih[u] + b_hh[u];
  const float* wi = w_ih + u * 2 * D;
#pragma unroll 8
  for (int j = 0; j < 2 * D; ++j) g += qs[j] * wi[j];
  const float* wh = w_hh + u * D;
#pragma unroll 8
  for (int j = 0; j < D; ++j) g += qs[2 * D + j] * wh[j];
  gates[b * 4 * D + u] = g;
}

__global__ void lstm_update(const float* __restrict__ gates, float* __restrict__ c,
                            float* __restrict__ h) {
  int g = blockIdx.x * 256 + threadIdx.x;
  if (g >= NG * D) return;
  int b = g >> 6, j = g & 63;
  const float* gr = gates + b * 4 * D;
  float ig = sigmoidf(gr[j]);
  float fg = sigmoidf(gr[D + j]);
  float gg = tanhf(gr[2 * D + j]);
  float og = sigmoidf(gr[3 * D + j]);
  float cn = fg * c[g] + ig * gg;
  c[g] = cn;
  h[g] = og * tanhf(cn);
}

__global__ void score_kernel(const float* __restrict__ xf, const float* __restrict__ h,
                             const int* __restrict__ batch, float* __restrict__ p,
                             unsigned* __restrict__ emax) {
  int i = blockIdx.x * 256 + threadIdx.x;
  if (i >= NN) return;
  int b = batch[i];
  const float* xr = xf + (size_t)i * D;
  const float* q = h + (size_t)b * D;
  float s = 0.f;
#pragma unroll 8
  for (int j = 0; j < D; ++j) s += xr[j] * q[j];
  p[i] = s;
  atomicMax(emax + b, fenc(s));
}

__global__ void exp_kernel(const int* __restrict__ batch, float* __restrict__ p,
                           const unsigned* __restrict__ emax, float* __restrict__ asum) {
  int i = blockIdx.x * 256 + threadIdx.x;
  if (i >= NN) return;
  int b = batch[i];
  float pe = expf(p[i] - fdec(emax[b]));
  p[i] = pe;
  atomAdd(asum + b, pe);
}

__global__ void accum_kernel(const float* __restrict__ xf, const int* __restrict__ batch,
                             const float* __restrict__ p, float* __restrict__ rtmp) {
  int g = blockIdx.x * 256 + threadIdx.x;
  if (g >= NN * D) return;
  int i = g >> 6, n = g & 63;
  atomAdd(&rtmp[(size_t)batch[i] * D + n], p[i] * xf[(size_t)i * D + n]);
}

__global__ void qstar_kernel(const float* __restrict__ h, const float* __restrict__ rtmp,
                             const float* __restrict__ asum, float* __restrict__ qstar) {
  int b = blockIdx.x, j = threadIdx.x;  // 128 threads
  float v = (j < D) ? h[b * D + j] : rtmp[b * D + (j - D)] / asum[b];
  qstar[b * 2 * D + j] = v;
}

// ---- head: out = relu(q_star@fc1+b)@fc4+b ---------------------------------
__global__ void head_kernel(const float* __restrict__ qstar, const float* __restrict__ fc1w,
                            const float* __restrict__ fc1b, const float* __restrict__ fc4w,
                            const float* __restrict__ fc4b, float* __restrict__ out) {
  __shared__ float hidv[D];
  int b = blockIdx.x, n = threadIdx.x;  // 64 threads
  const float* q = qstar + b * 2 * D;
  float s = fc1b[n];
#pragma unroll 8
  for (int j = 0; j < 2 * D; ++j) s += q[j] * fc1w[j * D + n];
  hidv[n] = s > 0.f ? s : 0.f;
  __syncthreads();
  if (n < NC) {
    float o = fc4b[n];
#pragma unroll 8
    for (int k = 0; k < D; ++k) o += hidv[k] * fc4w[k * NC + n];
    out[b * NC + n] = o;
  }
}

// ---------------------------------------------------------------------------
extern "C" void kernel_launch(void* const* d_in, const int* in_sizes, int n_in,
                              void* d_out, int out_size, void* d_ws, size_t ws_size,
                              hipStream_t stream) {
  (void)in_sizes; (void)n_in; (void)out_size; (void)ws_size;
  const float* x    = (const float*)d_in[0];
  const float* ea   = (const float*)d_in[1];
  const float* bw1  = (const float*)d_in[2];
  const float* bb1  = (const float*)d_in[3];
  const float* bw2  = (const float*)d_in[4];
  const float* bb2  = (const float*)d_in[5];
  const float* mw1  = (const float*)d_in[6];
  const float* mb1  = (const float*)d_in[7];
  const float* mw2  = (const float*)d_in[8];
  const float* mb2  = (const float*)d_in[9];
  const float* eps  = (const float*)d_in[10];
  const float* w_ih = (const float*)d_in[11];
  const float* w_hh = (const float*)d_in[12];
  const float* b_ih = (const float*)d_in[13];
  const float* b_hh = (const float*)d_in[14];
  const float* fc1w = (const float*)d_in[15];
  const float* fc1b = (const float*)d_in[16];
  const float* fc4w = (const float*)d_in[17];
  const float* fc4b = (const float*)d_in[18];
  const int* eidx   = (const int*)d_in[19];
  const int* batch  = (const int*)d_in[20];
  const int* srcI = eidx;
  const int* dstI = eidx + NE;

  // workspace carve-out (~40 MB), 256B aligned regions
  char* base = (char*)d_ws;
  auto carve = [&](size_t bytes) -> char* {
    char* p = base;
    base += (bytes + 255) & ~(size_t)255;
    return p;
  };
  float*     xa    = (float*)carve((size_t)NN * D * 4);
  float*     xb    = (float*)carve((size_t)NN * D * 4);
  float*     agg   = (float*)carve((size_t)NN * D * 4);
  _Float16*  wt    = (_Float16*)carve((size_t)NL * 4 * D * D * 2);
  float*     hbuf  = (float*)carve((size_t)NG * D * 4);
  float*     cbuf  = (float*)carve((size_t)NG * D * 4);
  float*     qstar = (float*)carve((size_t)NG * 2 * D * 4);
  float*     gates = (float*)carve((size_t)NG * 4 * D * 4);
  float*     pbuf  = (float*)carve((size_t)NN * 4);
  unsigned*  emax  = (unsigned*)carve((size_t)NG * 4);
  float*     asum  = (float*)carve((size_t)NG * 4);
  float*     rtmp  = (float*)carve((size_t)NG * D * 4);

  prep_weights<<<(NL * 4 * D * D + 255) / 256, 256, 0, stream>>>(bw1, bw2, mw1, mw2, wt);
  zero_f4<<<(NG * D / 4 + 255) / 256, 256, 0, stream>>>(hbuf, NG * D / 4);
  zero_f4<<<(NG * D / 4 + 255) / 256, 256, 0, stream>>>(cbuf, NG * D / 4);
  zero_f4<<<(NG * 2 * D / 4 + 255) / 256, 256, 0, stream>>>(qstar, NG * 2 * D / 4);

  const float* xin = x;
  float* xout = xa;
  for (int l = 0; l < NL; ++l) {
    zero_f4<<<(NN * D / 4 + 255) / 256, 256, 0, stream>>>(agg, NN * D / 4);
    edge_kernel<<<NE / 128, 256, 0, stream>>>(
        xin, ea, wt + (size_t)(l * 4 + 0) * D * D, bb1 + l * D,
        wt + (size_t)(l * 4 + 1) * D * D, bb2 + l * D, srcI, dstI, agg);
    node_kernel<<<(NODE_TILES + 7) / 8, 256, 0, stream>>>(
        xin, agg, eps, l, wt + (size_t)(l * 4 + 2) * D * D, mb1 + l * D,
        wt + (size_t)(l * 4 + 3) * D * D, mb2 + l * D, xout);
    xin = xout;
    xout = (xout == xa) ? xb : xa;
  }
  const float* xf = xin;  // final node features

  for (int t = 0; t < S2S; ++t) {
    gates_kernel<<<NG, 256, 0, stream>>>(qstar, hbuf, w_ih, w_hh, b_ih, b_hh, gates);
    lstm_update<<<(NG * D + 255) / 256, 256, 0, stream>>>(gates, cbuf, hbuf);
    zero_stats<<<(NG * D + 255) / 256, 256, 0, stream>>>(emax, asum, rtmp);
    score_kernel<<<(NN + 255) / 256, 256, 0, stream>>>(xf, hbuf, batch, pbuf, emax);
    exp_kernel<<<(NN + 255) / 256, 256, 0, stream>>>(batch, pbuf, emax, asum);
    accum_kernel<<<((size_t)NN * D + 255) / 256, 256, 0, stream>>>(xf, batch, pbuf, rtmp);
    qstar_kernel<<<NG, 128, 0, stream>>>(hbuf, rtmp, asum, qstar);
  }

  head_kernel<<<NG, D, 0, stream>>>(qstar, fc1w, fc1b, fc4w, fc4b, (float*)d_out);
}
